// VAE_56899726737506
// MI455X (gfx1250) — matrix operand order repaired and checked by reading, not compiled, not run on hardware
//
#include <hip/hip_runtime.h>
#include <hip/hip_bf16.h>

// ---- model dims ----
#define NB   32          // batch
#define NC   24
#define NT   1024
#define LL   24
#define DD   256         // Nt/4
#define C16  384
#define H8   192

typedef __attribute__((ext_vector_type(16))) _Float16 v16h;
typedef __attribute__((ext_vector_type(8)))  float    v8f;

#define DEV static __device__ __forceinline__

DEV float sigmf(float v){ return 1.0f/(1.0f+expf(-v)); }

// ---------------- WMMA fragment helpers (CDNA5 ISA 7.12.2 layouts) ----------------
// A: 16x32 f16.  lane&15 = M; VGPR v holds K = (v<4 ? 2v : 2v+8) + (lane>=16 ? 8 : 0), +1
template<typename F>
DEV v16h frag_a(F&& f, int lane){
  v16h a;
  const int m  = lane & 15;
  const int hi = (lane >> 4) & 1;
#pragma unroll
  for (int v = 0; v < 8; ++v){
    const int kb = ((v < 4) ? (2*v) : (2*v + 8)) + hi*8;
    a[2*v]   = (_Float16)f(m, kb);
    a[2*v+1] = (_Float16)f(m, kb + 1);
  }
  return a;
}
// B: 32x16 f16. lane&15 = N; VGPR v holds K = (v<4 ? 2v : 2v+8) + (lane>=16 ? 8 : 0), +1
template<typename F>
DEV v16h frag_b(F&& f, int lane){
  v16h b;
  const int n   = lane & 15;
  const int hi  = (lane >> 4) & 1;
#pragma unroll
  for (int v = 0; v < 8; ++v){
    const int kb = ((v < 4) ? (2*v) : (2*v + 8)) + hi*8;
    b[2*v]   = (_Float16)f(kb,     n);
    b[2*v+1] = (_Float16)f(kb + 1, n);
  }
  return b;
}
// C/D: 16x16 f32. lane&15 = N; VGPR r holds M = r + 8*(lane>=16)
template<typename F>
DEV v8f frag_c_load(F&& f, int lane){
  v8f c;
  const int n  = lane & 15;
  const int mo = ((lane >> 4) & 1) * 8;
#pragma unroll
  for (int r = 0; r < 8; ++r) c[r] = f(mo + r, n);
  return c;
}
template<typename F>
DEV void frag_c_store(const v8f& c, F&& f, int lane){
  const int n  = lane & 15;
  const int mo = ((lane >> 4) & 1) * 8;
#pragma unroll
  for (int r = 0; r < 8; ++r) f(mo + r, n, c[r]);
}
DEV v8f wmma16(v16h a, v16h b, v8f c){
  return __builtin_amdgcn_wmma_f32_16x16x32_f16(false, a, false, b, (short)0, c, false, false);
}
DEV int clampi(int v, int lo, int hi){ return v < lo ? lo : (v > hi ? hi : v); }

// ---------------- 1) sconv + bn1 + prelu : (B,1,24,1024) -> x1 (B,48,1024) ----------------
__global__ void k_sconv(const float* __restrict__ x, const float* __restrict__ w,
                        const float* __restrict__ bias,
                        const float* __restrict__ g, const float* __restrict__ be,
                        const float* __restrict__ mn, const float* __restrict__ vr,
                        const float* __restrict__ al, float* __restrict__ y)
{
  int idx = blockIdx.x * blockDim.x + threadIdx.x;
  if (idx >= NB*48*NT) return;
  int t = idx & (NT-1);
  int o = (idx >> 10) % 48;
  int b = idx / (48*NT);
  float acc = bias[o];
#pragma unroll
  for (int h = 0; h < NC; ++h)
    acc += x[(b*NC + h)*NT + t] * w[o*NC + h];
  float s  = g[o] * rsqrtf(vr[o] + 1e-5f);
  float yv = (acc - mn[o]) * s + be[o];
  y[idx] = yv >= 0.f ? yv : al[0]*yv;
}

// ---------------- 2) strided conv (k=16, stride 2, pad 7) + bn + prelu, WMMA ----------------
// out[b,o,t'] = prelu(bn( sum_{i,k} w[o,i,k]*in[b,i,2t'+k-7] + bias[o] ))
__global__ void k_tconv(const float* __restrict__ xin, const float* __restrict__ w,
                        const float* __restrict__ bias,
                        const float* __restrict__ g, const float* __restrict__ be,
                        const float* __restrict__ mn, const float* __restrict__ vr,
                        const float* __restrict__ al, float* __restrict__ y,
                        int Cin, int Cout, int Tin, int Tout)
{
  const int lane = threadIdx.x & 31;
  const int wave = threadIdx.x >> 5;
  const int ntT  = Tout >> 4;
  const int tile = blockIdx.x * (blockDim.x >> 5) + wave;
  if (tile >= (Cout >> 4) * NB * ntT) return;
  const int mt = tile / (NB * ntT);
  int rem = tile % (NB * ntT);
  const int b  = rem / ntT;
  const int t0 = (rem % ntT) << 4;
  const int m0 = mt << 4;
  const int K  = Cin << 4;
  v8f c = {};
  for (int k0 = 0; k0 < K; k0 += 32){
    if (k0 + 32 < K) __builtin_prefetch(&w[(m0)*K + k0 + 32], 0, 1);
    v16h A = frag_a([&](int m, int k){ return w[(m0 + m)*K + k0 + k]; }, lane);
    v16h Bm = frag_b([&](int k, int n){
      int kl = k0 + k;
      int ci = kl >> 4, kk = kl & 15;
      int t  = 2*(t0 + n) + kk - 7;
      int tc = clampi(t, 0, Tin - 1);
      float v = xin[((size_t)b*Cin + ci)*Tin + tc];   // unconditional, clamped
      return (t == tc) ? v : 0.0f;                    // select instead of exec-branch
    }, lane);
    c = wmma16(A, Bm, c);
  }
  frag_c_store(c, [&](int m, int n, float v){
    int o = m0 + m;
    float s  = g[o] * rsqrtf(vr[o] + 1e-5f);
    float yv = (v + bias[o] - mn[o]) * s + be[o];
    y[((size_t)b*Cout + o)*Tout + t0 + n] = yv >= 0.f ? yv : al[0]*yv;
  }, lane);
}

// ---------------- 3) transpose x3 (B,192,256) -> xts (T*B, 192) ----------------
__global__ void k_xts(const float* __restrict__ x3, float* __restrict__ xts)
{
  int idx = blockIdx.x*blockDim.x + threadIdx.x;
  if (idx >= DD*NB*H8) return;
  int i = idx % H8;
  int m = idx / H8;          // t*NB + b
  int t = m / NB, b = m % NB;
  xts[idx] = x3[((size_t)b*H8 + i)*DD + t];
}

// ---------------- 4) LSTM input projection: gx = xts @ w_ih^T + (b_ih+b_hh), WMMA ----------------
__global__ void k_gx(const float* __restrict__ xts, const float* __restrict__ wih,
                     const float* __restrict__ bih, const float* __restrict__ bhh,
                     float* __restrict__ gx)
{
  const int lane = threadIdx.x & 31;
  const int wave = threadIdx.x >> 5;
  const int tile = blockIdx.x * (blockDim.x >> 5) + wave;   // 512*48 tiles
  const int mt = tile / 48, nt = tile % 48;
  const int m0 = mt << 4, n0 = nt << 4;
  v8f c = {};
  for (int k0 = 0; k0 < H8; k0 += 32){
    v16h A = frag_a([&](int m, int k){ return xts[(size_t)(m0 + m)*H8 + k0 + k]; }, lane);
    v16h Bm = frag_b([&](int k, int n){ return wih[(size_t)(n0 + n)*H8 + k0 + k]; }, lane);
    c = wmma16(A, Bm, c);
  }
  frag_c_store(c, [&](int m, int n, float v){
    int gi = n0 + n;
    gx[(size_t)(m0 + m)*768 + gi] = v + bih[gi] + bhh[gi];
  }, lane);
}

// ---------------- 5) LSTM recurrence: one persistent block per direction, WMMA ----------------
__global__ void k_lstm(const float* __restrict__ gx_f, const float* __restrict__ gx_b,
                       const float* __restrict__ whh_f, const float* __restrict__ whh_b,
                       float* __restrict__ hf, float* __restrict__ hb)
{
  extern __shared__ float sm[];
  float* hS = sm;            // 32*192 = 6144
  float* cS = sm + 6144;     // 6144
  float* gS = sm + 12288;    // 32*768 = 24576
  const int dir = blockIdx.x;
  const float* gx  = dir ? gx_b  : gx_f;
  const float* whh = dir ? whh_b : whh_f;
  float* hs = dir ? hb : hf;
  for (int i = threadIdx.x; i < 6144; i += blockDim.x){ hS[i] = 0.f; cS[i] = 0.f; }
  __syncthreads();
  const int lane = threadIdx.x & 31;
  const int wave = threadIdx.x >> 5;     // 16 waves
  for (int step = 0; step < DD; ++step){
    const int t = dir ? (DD - 1 - step) : step;
    // gates(32x768) = gx[t] + hS(32x192) @ whh^T : 96 tiles over 16 waves
#pragma unroll
    for (int ti = 0; ti < 6; ++ti){
      const int tile = wave*6 + ti;
      const int mt = tile / 48, nt = tile % 48;
      const int m0 = mt << 4, n0 = nt << 4;
      v8f c = frag_c_load([&](int m, int n){
        return gx[(size_t)(t*NB + m0 + m)*768 + n0 + n]; }, lane);
      for (int k0 = 0; k0 < H8; k0 += 32){
        __builtin_prefetch(&whh[(size_t)(n0)*H8 + ((k0 + 32) % H8)], 0, 1);
        v16h A = frag_a([&](int m, int k){ return hS[(m0 + m)*H8 + k0 + k]; }, lane);
        v16h Bm = frag_b([&](int k, int n){ return whh[(size_t)(n0 + n)*H8 + k0 + k]; }, lane);
        c = wmma16(A, Bm, c);
      }
      frag_c_store(c, [&](int m, int n, float v){ gS[(m0 + m)*768 + n0 + n] = v; }, lane);
    }
    __syncthreads();
    // gate nonlinearities (i,f,g,o) + state update
    for (int e = threadIdx.x; e < 6144; e += blockDim.x){
      int b = e / H8, j = e % H8;
      float gi = gS[b*768 + j];
      float gf = gS[b*768 + H8 + j];
      float gg = gS[b*768 + 2*H8 + j];
      float go = gS[b*768 + 3*H8 + j];
      float cc = sigmf(gf)*cS[e] + sigmf(gi)*tanhf(gg);
      float hh = sigmf(go)*tanhf(cc);
      cS[e] = cc; hS[e] = hh;
      hs[(size_t)(t*NB + b)*H8 + j] = hh;
    }
    __syncthreads();
  }
}

// ---------------- 6) avg-pool(2) over features + transpose + concat -> e0 (B,384,256) ----------------
__global__ void k_e0(const float* __restrict__ hf, const float* __restrict__ hb,
                     const float* __restrict__ x3, float* __restrict__ e0)
{
  int idx = blockIdx.x*blockDim.x + threadIdx.x;
  if (idx >= NB*C16*DD) return;
  int t = idx & (DD-1);
  int c = (idx >> 8) % C16;
  int b = idx / (C16*DD);
  float v;
  if (c < 96){
    int j = 2*c;
    const float* h = hf + (size_t)(t*NB + b)*H8;
    v = 0.5f*(h[j] + h[j+1]);
  } else if (c < 192){
    int j = 2*(c - 96);
    const float* h = hb + (size_t)(t*NB + b)*H8;
    v = 0.5f*(h[j] + h[j+1]);
  } else {
    v = x3[((size_t)b*H8 + (c - 192))*DD + t];
  }
  e0[idx] = v;
}

// ---------------- 7) per-channel mu/logvar linears + reparam z, WMMA (shared A) ----------------
__global__ void k_mulvz(const float* __restrict__ e0, const float* __restrict__ eps,
                        const float* __restrict__ mw, const float* __restrict__ mb,
                        const float* __restrict__ lw, const float* __restrict__ lb,
                        float* __restrict__ mu, float* __restrict__ lv, float* __restrict__ z)
{
  const int c = blockIdx.x;               // 384 channels
  const int lane = threadIdx.x & 31;
  const int wave = threadIdx.x >> 5;      // 4 waves: M(2) x N(2) tiles
  const int m0 = (wave & 1) << 4;
  const int n0 = (wave >> 1) << 4;
  v8f cm = {}, cl = {};
  for (int k0 = 0; k0 < DD; k0 += 32){
    v16h A = frag_a([&](int m, int k){
      return e0[((size_t)(m0 + m)*C16 + c)*DD + k0 + k]; }, lane);
    v16h Bm = frag_b([&](int k, int n){
      int l = n0 + n;
      int lc = clampi(l, 0, LL - 1);
      float v = mw[((size_t)c*LL + lc)*DD + k0 + k];
      return (l == lc) ? v : 0.0f; }, lane);
    v16h Bl = frag_b([&](int k, int n){
      int l = n0 + n;
      int lc = clampi(l, 0, LL - 1);
      float v = lw[((size_t)c*LL + lc)*DD + k0 + k];
      return (l == lc) ? v : 0.0f; }, lane);
    cm = wmma16(A, Bm, cm);
    cl = wmma16(A, Bl, cl);
  }
  const int n  = lane & 15;
  const int mo = ((lane >> 4) & 1) * 8;
#pragma unroll
  for (int r = 0; r < 8; ++r){
    int l = n0 + n;
    if (l < LL){
      int b = m0 + mo + r;
      float m_ = cm[r] + mb[c*LL + l];
      float v_ = cl[r] + lb[c*LL + l];
      size_t oi = ((size_t)b*C16 + c)*LL + l;
      mu[oi] = m_;
      lv[oi] = v_;
      z[oi]  = m_ + eps[oi]*expf(0.5f*v_);
    }
  }
}

// ---------------- 8) per-channel decoder linears: x4 = z @ dec_w^T + dec_b, WMMA ----------------
__global__ void k_dec(const float* __restrict__ z, const float* __restrict__ dw,
                      const float* __restrict__ db, float* __restrict__ x4)
{
  const int c = blockIdx.x;
  const int lane = threadIdx.x & 31;
  const int wave = threadIdx.x >> 5;      // 8 waves x 4 tiles = 32 tiles (M2 x N16)
#pragma unroll
  for (int s = 0; s < 4; ++s){
    const int tile = wave*4 + s;
    const int m0 = (tile >> 4) << 4;
    const int n0 = (tile & 15) << 4;
    v8f cc = {};
    v16h A = frag_a([&](int m, int k){
      int kc = clampi(k, 0, LL - 1);
      float v = z[((size_t)(m0 + m)*C16 + c)*LL + kc];
      return (k == kc) ? v : 0.0f; }, lane);
    v16h Bm = frag_b([&](int k, int n){
      int kc = clampi(k, 0, LL - 1);
      float v = dw[((size_t)c*DD + n0 + n)*LL + kc];
      return (k == kc) ? v : 0.0f; }, lane);
    cc = wmma16(A, Bm, cc);
    frag_c_store(cc, [&](int m, int n, float v){
      int d = n0 + n;
      x4[((size_t)(m0 + m)*C16 + c)*DD + d] = v + db[c*DD + d];
    }, lane);
  }
}

// ---------------- 9) ConvTranspose (k=16, stride 2, pad 7) + bn + prelu, parity split, WMMA ----
// out[b,o,2t'+p] = prelu(bn( sum_{i,j} w[i,o,2j+koff] * in[b,i,t'+toff-j] + bias[o] ))
__global__ void k_dconv(const float* __restrict__ s0, const float* __restrict__ s1,
                        int C0, int C1,
                        const float* __restrict__ w, const float* __restrict__ bias,
                        const float* __restrict__ g, const float* __restrict__ be,
                        const float* __restrict__ mn, const float* __restrict__ vr,
                        const float* __restrict__ al, float* __restrict__ y,
                        int Cin, int Cout, int Tin)
{
  const int lane = threadIdx.x & 31;
  const int wave = threadIdx.x >> 5;
  const int ntT  = Tin >> 4;
  int tile = blockIdx.x * (blockDim.x >> 5) + wave;
  if (tile >= (Cout >> 4) * NB * 2 * ntT) return;
  const int mt = tile / (NB*2*ntT);
  int rem = tile % (NB*2*ntT);
  const int b = rem / (2*ntT); rem %= 2*ntT;
  const int p = rem / ntT;
  const int t0 = (rem % ntT) << 4;
  const int m0 = mt << 4;
  const int koff = p ? 0 : 1;
  const int toff = p ? 4 : 3;
  const int K = Cin << 3;
  v8f c = {};
  for (int k0 = 0; k0 < K; k0 += 32){
    if (k0 + 32 < K) __builtin_prefetch(&w[(size_t)(((k0+32) >> 3)*Cout + m0) << 4], 0, 1);
    v16h A = frag_a([&](int m, int kc){
      int col = k0 + kc; int ci = col >> 3, j = col & 7;
      return w[((size_t)(ci*Cout + m0 + m) << 4) + 2*j + koff]; }, lane);
    v16h Bm = frag_b([&](int kc, int n){
      int col = k0 + kc; int ci = col >> 3, j = col & 7;
      int tin = t0 + n + toff - j;
      int tc  = clampi(tin, 0, Tin - 1);
      float v = (ci < C0) ? s0[((size_t)b*C0 + ci)*Tin + tc]
                          : s1[((size_t)b*C1 + (ci - C0))*Tin + tc];
      return (tin == tc) ? v : 0.0f; }, lane);
    c = wmma16(A, Bm, c);
  }
  const int Tout = Tin << 1;
  frag_c_store(c, [&](int m, int n, float v){
    int o = m0 + m;
    float s  = g[o]*rsqrtf(vr[o] + 1e-5f);
    float yv = (v + bias[o] - mn[o])*s + be[o];
    yv = yv >= 0.f ? yv : al[0]*yv;
    y[((size_t)b*Cout + o)*Tout + 2*(t0 + n) + p] = yv;
  }, lane);
}

// ---------------- 10) fused sdconvT + bn6 + prelu + fconv -> out (B,1,24,1024) ----------------
__global__ void k_final(const float* __restrict__ u2, const float* __restrict__ x1,
                        const float* __restrict__ sw, const float* __restrict__ sb,
                        const float* __restrict__ g, const float* __restrict__ be,
                        const float* __restrict__ mn, const float* __restrict__ vr,
                        const float* __restrict__ al,
                        const float* __restrict__ fw, const float* __restrict__ fb,
                        float* __restrict__ out)
{
  int idx = blockIdx.x*blockDim.x + threadIdx.x;
  if (idx >= NB*NC*NT) return;
  int t = idx & (NT-1);
  int h = (idx >> 10) % NC;
  int b = idx / (NC*NT);
  float acc[48];
#pragma unroll
  for (int o = 0; o < 48; ++o) acc[o] = sb[o];
  for (int i = 0; i < 96; ++i){
    float e = (i < 48) ? u2[((size_t)b*48 + i)*NT + t]
                       : x1[((size_t)b*48 + (i - 48))*NT + t];
#pragma unroll
    for (int o = 0; o < 48; ++o) acc[o] += e * sw[(i*48 + o)*NC + h];
  }
  float r = fb[0];
#pragma unroll
  for (int o = 0; o < 48; ++o){
    float s  = g[o]*rsqrtf(vr[o] + 1e-5f);
    float yv = (acc[o] - mn[o])*s + be[o];
    yv = yv >= 0.f ? yv : al[0]*yv;
    r += fw[o]*yv;
  }
  out[idx] = r;
}

// =====================================================================================
extern "C" void kernel_launch(void* const* d_in, const int* in_sizes, int n_in,
                              void* d_out, int out_size, void* d_ws, size_t ws_size,
                              hipStream_t stream)
{
  (void)in_sizes; (void)n_in; (void)out_size; (void)ws_size;
  const float* x        = (const float*)d_in[0];
  const float* eps      = (const float*)d_in[1];
  const float* sconv_w  = (const float*)d_in[2];
  const float* sconv_b  = (const float*)d_in[3];
  const float* tconv1_w = (const float*)d_in[4];
  const float* tconv1_b = (const float*)d_in[5];
  const float* tconv2_w = (const float*)d_in[6];
  const float* tconv2_b = (const float*)d_in[7];
  const float *bng[6], *bnb[6], *bnm[6], *bnv[6], *aa[6];
  for (int i = 0; i < 6; ++i){
    bng[i] = (const float*)d_in[8  + i*5];
    bnb[i] = (const float*)d_in[9  + i*5];
    bnm[i] = (const float*)d_in[10 + i*5];
    bnv[i] = (const float*)d_in[11 + i*5];
    aa[i]  = (const float*)d_in[12 + i*5];
  }
  const float* w_ih_f = (const float*)d_in[38];
  const float* w_hh_f = (const float*)d_in[39];
  const float* b_ih_f = (const float*)d_in[40];
  const float* b_hh_f = (const float*)d_in[41];
  const float* w_ih_b = (const float*)d_in[42];
  const float* w_hh_b = (const float*)d_in[43];
  const float* b_ih_b = (const float*)d_in[44];
  const float* b_hh_b = (const float*)d_in[45];
  const float* mu_w   = (const float*)d_in[46];
  const float* mu_b   = (const float*)d_in[47];
  const float* lv_w   = (const float*)d_in[48];
  const float* lv_b   = (const float*)d_in[49];
  const float* dec_w  = (const float*)d_in[50];
  const float* dec_b  = (const float*)d_in[51];
  const float* dconv1_w = (const float*)d_in[52];
  const float* dconv1_b = (const float*)d_in[53];
  const float* dconv2_w = (const float*)d_in[54];
  const float* dconv2_b = (const float*)d_in[55];
  const float* sdconv_w = (const float*)d_in[56];
  const float* sdconv_b = (const float*)d_in[57];
  const float* fconv_w  = (const float*)d_in[58];
  const float* fconv_b  = (const float*)d_in[59];

  float* ws  = (float*)d_ws;
  float* x1  = ws + 0;                 // 1572864
  float* x2  = ws + 1572864;           // 1572864
  float* x3  = ws + 3145728;           // 1572864
  float* xts = ws + 4718592;           // 1572864
  float* gxf = ws + 6291456;           // 6291456
  float* gxb = ws + 12582912;          // 6291456
  float* hf  = ws + 18874368;          // 1572864
  float* hb  = ws + 20447232;          // 1572864  (peak ~88 MB)
  // aliases valid after the LSTM phase (stream-ordered):
  float* e0  = gxf;                    // 3145728 (gx_f dead)
  float* z   = xts;                    // 294912  (xts dead)
  float* x4  = gxb;                    // 3145728 (gx_b dead)
  float* u1  = gxb + 3145728;          // 1572864 (fits inside gx_b region)
  float* u2  = hf;                     // 1572864 (hf dead after e0)

  float* outp = (float*)d_out;         // (B,1,24,1024) = 786432
  float* muo  = outp + NB*NC*NT;       // (B,384,24)    = 294912
  float* lvo  = muo + NB*C16*LL;       // (B,384,24)    = 294912

  // Encoder
  k_sconv<<<6144, 256, 0, stream>>>(x, sconv_w, sconv_b, bng[0], bnb[0], bnm[0], bnv[0], aa[0], x1);
  k_tconv<<<768, 256, 0, stream>>>(x1, tconv1_w, tconv1_b, bng[1], bnb[1], bnm[1], bnv[1], aa[1],
                                   x2, 48, 96, 1024, 512);
  k_tconv<<<768, 256, 0, stream>>>(x2, tconv2_w, tconv2_b, bng[2], bnb[2], bnm[2], bnv[2], aa[2],
                                   x3, 96, 192, 512, 256);
  // BiLSTM
  k_xts<<<6144, 256, 0, stream>>>(x3, xts);
  k_gx<<<3072, 256, 0, stream>>>(xts, w_ih_f, b_ih_f, b_hh_f, gxf);
  k_gx<<<3072, 256, 0, stream>>>(xts, w_ih_b, b_ih_b, b_hh_b, gxb);
  k_lstm<<<2, 512, 147456, stream>>>(gxf, gxb, w_hh_f, w_hh_b, hf, hb);
  k_e0<<<12288, 256, 0, stream>>>(hf, hb, x3, e0);
  // VAE heads
  k_mulvz<<<384, 128, 0, stream>>>(e0, eps, mu_w, mu_b, lv_w, lv_b, muo, lvo, z);
  k_dec<<<384, 256, 0, stream>>>(z, dec_w, dec_b, x4);
  // Decoder
  k_dconv<<<768, 256, 0, stream>>>(x4, x4, 384, 0, dconv1_w, dconv1_b,
                                   bng[3], bnb[3], bnm[3], bnv[3], aa[3], u1, 384, 96, 256);
  k_dconv<<<768, 256, 0, stream>>>(u1, x2, 96, 96, dconv2_w, dconv2_b,
                                   bng[4], bnb[4], bnm[4], bnv[4], aa[4], u2, 192, 48, 512);
  k_final<<<3072, 256, 0, stream>>>(u2, x1, sdconv_w, sdconv_b,
                                    bng[5], bnb[5], bnm[5], bnv[5], aa[5],
                                    fconv_w, fconv_b, outp);
}
